// MMD_42984032698619
// MI455X (gfx1250) — compile-verified, matrix operand order checked by reading, not run on hardware
//
#include <hip/hip_runtime.h>

// ---------------------------------------------------------------------------
// MMD with multi-gamma Gaussian kernels on MI455X (gfx1250, wave32, WMMA).
// Split-bf16 GEMM (hi*hi + hi*lo + lo*hi) via v_wmma_f32_16x16x32_bf16 with
// 2x2 register blocking (32x32 outputs per wave), fused norm/exp epilogue,
// deterministic two-stage reduction.
// ---------------------------------------------------------------------------

typedef __attribute__((ext_vector_type(8)))  __bf16 v8bf;
typedef __attribute__((ext_vector_type(16))) __bf16 v16bf;
typedef __attribute__((ext_vector_type(8)))  float  v8f;

static constexpr int NROWS = 8192;
static constexpr int DDIM  = 256;
static constexpr int WT    = NROWS / 32;        // 256 wave-tiles (32x32) per dim
static constexpr int GEMM_BLOCKS = WT * (WT / 8); // 256 * 32 = 8192 blocks/pass

#if __has_builtin(__builtin_amdgcn_exp2f)
#define EXP2F(x) __builtin_amdgcn_exp2f(x)
#else
#define EXP2F(x) exp2f(x)
#endif

__device__ __forceinline__ unsigned short f32_to_bf16_rne(float f) {
  unsigned u = __builtin_bit_cast(unsigned, f);
  unsigned r = u + 0x7FFFu + ((u >> 16) & 1u);   // round-to-nearest-even
  return (unsigned short)(r >> 16);
}
__device__ __forceinline__ float bf16_to_f32(unsigned short s) {
  unsigned u = ((unsigned)s) << 16;
  return __builtin_bit_cast(float, u);
}

// Load one 16x32 bf16 WMMA fragment slice for this lane: two contiguous
// 16-byte runs at p and p+16 elements (matches the ISA 16-bit A/B layout:
// v16bf elements 0..7 <- K=k0..k0+7, elements 8..15 <- K=k0+16..k0+23).
__device__ __forceinline__ v16bf load_frag(const unsigned short* __restrict__ p) {
  v8bf lo = *(const v8bf*)(p);
  v8bf hi = *(const v8bf*)(p + 16);
  return __builtin_shufflevector(lo, hi, 0,1,2,3,4,5,6,7,8,9,10,11,12,13,14,15);
}

// Multi-gamma Gaussian kernel value: sum_g exp(-g*d) == sum_g exp2(cg*d).
__device__ __forceinline__ float gauss7(float d) {
  constexpr float L2E = 1.44269504088896340736f;
  d = fmaxf(d, 1e-30f);
  float k;
  k  = EXP2F(d * (-0.001f  * L2E));
  k += EXP2F(d * (-0.01f   * L2E));
  k += EXP2F(d * (-0.1f    * L2E));
  k += EXP2F(d * (-1.0f    * L2E));
  k += EXP2F(d * (-10.0f   * L2E));
  k += EXP2F(d * (-100.0f  * L2E));
  k += EXP2F(d * (-1000.0f * L2E));
  return k;
}

// ---------------------------------------------------------------------------
// Kernel 1: per-row bf16 hi/lo split + exact fp32 row norms.
// One block (256 threads) per row of 256 elements.
// ---------------------------------------------------------------------------
__global__ __launch_bounds__(256) void prep_kernel(
    const float* __restrict__ src,
    unsigned short* __restrict__ hi,
    unsigned short* __restrict__ lo,
    float* __restrict__ norm) {
  __shared__ float red[256];
  const int r = blockIdx.x;
  const int t = threadIdx.x;
  const float v = src[r * DDIM + t];
  const unsigned short h = f32_to_bf16_rne(v);
  const float hf = bf16_to_f32(h);
  const unsigned short l = f32_to_bf16_rne(v - hf);
  hi[r * DDIM + t] = h;
  lo[r * DDIM + t] = l;
  red[t] = v * v;
  __syncthreads();
  #pragma unroll
  for (int s = 128; s > 0; s >>= 1) {
    if (t < s) red[t] += red[t + s];
    __syncthreads();
  }
  if (t == 0) norm[r] = red[0];
}

// ---------------------------------------------------------------------------
// Kernel 2: tiled GEMM (A x B^T) with split-bf16 WMMA, 2x2 tile blocking
// (32x32 outputs per wave => 12 WMMAs per 16 b128 loads per k-chunk), and
// fused Gaussian-kernel epilogue. 8 waves (256 threads) per block.
// Block b: wave-tile row wtm = b>>5, col wtn = (b&31)*8 + wave_id.
// ---------------------------------------------------------------------------
__global__ __launch_bounds__(256) void mmd_gemm_kernel(
    const unsigned short* __restrict__ Ahi, const unsigned short* __restrict__ Alo,
    const float* __restrict__ An,
    const unsigned short* __restrict__ Bhi, const unsigned short* __restrict__ Blo,
    const float* __restrict__ Bn,
    float* __restrict__ partials, float coef) {
  const int lane = threadIdx.x & 31;
  const int wid  = threadIdx.x >> 5;
  const int wtm  = blockIdx.x >> 5;             // 0..255 (32-row strip)
  const int wtn  = (blockIdx.x & 31) * 8 + wid; // 0..255 (32-col strip)

  const int kb  = (lane >> 4) * 8;
  const int ar0 = wtm * 32 + (lane & 15);       // A fragment rows, tile 0
  const int br0 = wtn * 32 + (lane & 15);       // B fragment rows (B = rows of
                                                // the source matrix, A x B^T)
  const unsigned short* __restrict__ pa0h = Ahi + ar0 * DDIM;
  const unsigned short* __restrict__ pa0l = Alo + ar0 * DDIM;
  const unsigned short* __restrict__ pa1h = pa0h + 16 * DDIM;
  const unsigned short* __restrict__ pa1l = pa0l + 16 * DDIM;
  const unsigned short* __restrict__ pb0h = Bhi + br0 * DDIM;
  const unsigned short* __restrict__ pb0l = Blo + br0 * DDIM;
  const unsigned short* __restrict__ pb1h = pb0h + 16 * DDIM;
  const unsigned short* __restrict__ pb1l = pb0l + 16 * DDIM;

  v8f acc00 = {}, acc01 = {}, acc10 = {}, acc11 = {};
  #pragma unroll
  for (int kc = 0; kc < DDIM; kc += 32) {
    const int k0 = kc + kb;
    v16bf a0h = load_frag(pa0h + k0);
    v16bf a0l = load_frag(pa0l + k0);
    v16bf a1h = load_frag(pa1h + k0);
    v16bf a1l = load_frag(pa1l + k0);
    v16bf b0h = load_frag(pb0h + k0);
    v16bf b0l = load_frag(pb0l + k0);
    v16bf b1h = load_frag(pb1h + k0);
    v16bf b1l = load_frag(pb1l + k0);
    // s += hi*hi + hi*lo + lo*hi   (lo*lo term ~2^-16 relative, dropped)
    acc00 = __builtin_amdgcn_wmma_f32_16x16x32_bf16(false, a0h, false, b0h, (short)0, acc00, false, false);
    acc00 = __builtin_amdgcn_wmma_f32_16x16x32_bf16(false, a0h, false, b0l, (short)0, acc00, false, false);
    acc00 = __builtin_amdgcn_wmma_f32_16x16x32_bf16(false, a0l, false, b0h, (short)0, acc00, false, false);
    acc01 = __builtin_amdgcn_wmma_f32_16x16x32_bf16(false, a0h, false, b1h, (short)0, acc01, false, false);
    acc01 = __builtin_amdgcn_wmma_f32_16x16x32_bf16(false, a0h, false, b1l, (short)0, acc01, false, false);
    acc01 = __builtin_amdgcn_wmma_f32_16x16x32_bf16(false, a0l, false, b1h, (short)0, acc01, false, false);
    acc10 = __builtin_amdgcn_wmma_f32_16x16x32_bf16(false, a1h, false, b0h, (short)0, acc10, false, false);
    acc10 = __builtin_amdgcn_wmma_f32_16x16x32_bf16(false, a1h, false, b0l, (short)0, acc10, false, false);
    acc10 = __builtin_amdgcn_wmma_f32_16x16x32_bf16(false, a1l, false, b0h, (short)0, acc10, false, false);
    acc11 = __builtin_amdgcn_wmma_f32_16x16x32_bf16(false, a1h, false, b1h, (short)0, acc11, false, false);
    acc11 = __builtin_amdgcn_wmma_f32_16x16x32_bf16(false, a1h, false, b1l, (short)0, acc11, false, false);
    acc11 = __builtin_amdgcn_wmma_f32_16x16x32_bf16(false, a1l, false, b1h, (short)0, acc11, false, false);
  }

  // Epilogue. C/D layout: VGPR v holds M = v + 8*(lane>=16), N = lane&15.
  const int   cb  = wtn * 32 + (lane & 15);
  const float bn0 = Bn[cb];
  const float bn1 = Bn[cb + 16];
  const int   rb  = wtm * 32 + ((lane >> 4) << 3);
  float psum = 0.0f;
  #pragma unroll
  for (int v = 0; v < 8; ++v) {
    const float a0 = An[rb + v];
    const float a1 = An[rb + 16 + v];
    psum += gauss7(a0 + bn0 - 2.0f * acc00[v]);
    psum += gauss7(a0 + bn1 - 2.0f * acc01[v]);
    psum += gauss7(a1 + bn0 - 2.0f * acc10[v]);
    psum += gauss7(a1 + bn1 - 2.0f * acc11[v]);
  }

  // Fixed-order block reduction -> one partial per block (deterministic,
  // no float atomics so graph replays are bit-identical).
  __shared__ float red[256];
  red[threadIdx.x] = psum;
  __syncthreads();
  #pragma unroll
  for (int s = 128; s > 0; s >>= 1) {
    if (threadIdx.x < s) red[threadIdx.x] += red[threadIdx.x + s];
    __syncthreads();
  }
  if (threadIdx.x == 0) partials[blockIdx.x] = red[0] * coef;
}

// ---------------------------------------------------------------------------
// Kernel 3: deterministic final reduction of all partials into d_out[0].
// ---------------------------------------------------------------------------
__global__ __launch_bounds__(256) void finalize_kernel(
    const float* __restrict__ partials, int n, float* __restrict__ out) {
  __shared__ float red[256];
  const int t = threadIdx.x;
  float s = 0.0f;
  for (int i = t; i < n; i += 256) s += partials[i];
  red[t] = s;
  __syncthreads();
  #pragma unroll
  for (int k = 128; k > 0; k >>= 1) {
    if (t < k) red[t] += red[t + k];
    __syncthreads();
  }
  if (t == 0) out[0] = red[0];
}

// ---------------------------------------------------------------------------
// Launch: prep(x), prep(y), three GEMM passes (xx, yy, xy with coef -2),
// finalize. Workspace layout (bytes):
//   [0)        x_hi   : 8192*256*2 = 4 MiB
//   [4 MiB)    x_lo   : 4 MiB
//   [8 MiB)    y_hi   : 4 MiB
//   [12 MiB)   y_lo   : 4 MiB
//   [16 MiB)   x_norm : 8192*4
//   [+32KiB)   y_norm : 8192*4
//   [+64KiB)   partials: 3*8192*4
// ---------------------------------------------------------------------------
extern "C" void kernel_launch(void* const* d_in, const int* in_sizes, int n_in,
                              void* d_out, int out_size, void* d_ws, size_t ws_size,
                              hipStream_t stream) {
  (void)in_sizes; (void)n_in; (void)out_size; (void)ws_size;
  const float* x = (const float*)d_in[0];
  const float* y = (const float*)d_in[1];
  float* out = (float*)d_out;

  char* ws = (char*)d_ws;
  const size_t BF = (size_t)NROWS * DDIM * sizeof(unsigned short); // 4 MiB
  unsigned short* xhi = (unsigned short*)(ws);
  unsigned short* xlo = (unsigned short*)(ws + BF);
  unsigned short* yhi = (unsigned short*)(ws + 2 * BF);
  unsigned short* ylo = (unsigned short*)(ws + 3 * BF);
  float* xn = (float*)(ws + 4 * BF);
  float* yn = (float*)(ws + 4 * BF + 32768);
  float* partials = (float*)(ws + 4 * BF + 65536);

  prep_kernel<<<NROWS, 256, 0, stream>>>(x, xhi, xlo, xn);
  prep_kernel<<<NROWS, 256, 0, stream>>>(y, yhi, ylo, yn);

  const float inv = 1.0f / ((float)NROWS * (float)NROWS);
  // kxx + kyy - 2*kxy, each term is a mean over N*M entries.
  mmd_gemm_kernel<<<GEMM_BLOCKS, 256, 0, stream>>>(
      xhi, xlo, xn, xhi, xlo, xn, partials + 0 * GEMM_BLOCKS,  1.0f * inv);
  mmd_gemm_kernel<<<GEMM_BLOCKS, 256, 0, stream>>>(
      yhi, ylo, yn, yhi, ylo, yn, partials + 1 * GEMM_BLOCKS,  1.0f * inv);
  mmd_gemm_kernel<<<GEMM_BLOCKS, 256, 0, stream>>>(
      xhi, xlo, xn, yhi, ylo, yn, partials + 2 * GEMM_BLOCKS, -2.0f * inv);

  finalize_kernel<<<1, 256, 0, stream>>>(partials, 3 * GEMM_BLOCKS, out);
}
